// TrainableMMSEModule_32727650796135
// MI455X (gfx1250) — matrix operand order repaired and checked by reading, not compiled, not run on hardware
//
#include <hip/hip_runtime.h>

// ---------------- CDNA5 WMMA types ----------------
typedef __attribute__((ext_vector_type(16))) _Float16 v16h;
typedef __attribute__((ext_vector_type(8)))  float    v8f;

#define THREADS 256
#define ROWS_WG 128            // chunks per workgroup (8 waves x 16 rows)
#define SAH 136                // packed-f16 activation row stride (halves): 16B aligned, bank-skewed
#define FS 148                 // factor LDS row stride (floats), bank-skewed

// K-major pair-packed f16 weights in d_ws / LDS: wT[n][kp2], u32 = {half(2k+1),half(2k)}
// row stride padded (+4 u32) so b128 loads from 16 lanes hit distinct banks.
#define WS_L0 20               // K=32  -> 16 pairs (+4 pad)
#define WS_L1 68               // K=128 -> 64 pairs (+4 pad)
#define WS_L2 68
#define WS_L3 36               // K=64  -> 32 pairs (+4 pad)
#define W_L0_OFF 0
#define W_L0_SZ  (128*WS_L0)   // 2560
#define W_L1_OFF (W_L0_OFF + W_L0_SZ)
#define W_L1_SZ  (128*WS_L1)   // 8704
#define W_L2_OFF (W_L1_OFF + W_L1_SZ)
#define W_L2_SZ  (64*WS_L2)    // 4352
#define W_L3_OFF (W_L2_OFF + W_L2_SZ)
#define W_L3_SZ  (144*WS_L3)   // 5184
#define W_TOT    (W_L3_OFF + W_L3_SZ)   // 20800 u32 per MLP

// dynamic LDS layout (bytes)
#define LDS_W_B   0
#define LDS_ACT_B (W_TOT*4)                        // 83200
#define LDS_FC_B  (LDS_ACT_B + 8*16*SAH*2)         // +34816 = 118016
#define LDS_FR_B  (LDS_FC_B  + ROWS_WG*FS*4)       // +75776 = 193792
#define LDS_PRM_B (LDS_FR_B  + ROWS_WG*FS*4)       // +75776 = 269568
#define LDS_TOT_B (LDS_PRM_B + 4096)               // = 273664 (< 320KB WGP LDS)

static __device__ __forceinline__ unsigned short f2h_bits(float f) {
    _Float16 h = (_Float16)f;
    return __builtin_bit_cast(unsigned short, h);
}

// Same-wave LDS ordering is guaranteed by HW (DScnt in-order); only stop compiler reordering.
__device__ __forceinline__ void lds_fence() { asm volatile("" ::: "memory"); }

// ---------------- weight pack kernel: fp32 [K][N] -> K-major padded pair-packed f16 ----------------
__global__ void pack_weights(const float* __restrict__ w, int Kreal, int strideU, int N,
                             unsigned* __restrict__ out) {
    int idx = blockIdx.x * blockDim.x + threadIdx.x;
    int total = N * strideU;
    if (idx >= total) return;
    int n = idx / strideU, kp2 = idx % strideU;
    int k0 = 2 * kp2, k1 = k0 + 1;
    float f0 = (k0 < Kreal) ? w[k0 * N + n] : 0.f;
    float f1 = (k1 < Kreal) ? w[k1 * N + n] : 0.f;
    out[idx] = (unsigned)f2h_bits(f0) | ((unsigned)f2h_bits(f1) << 16);
}

// ---------------- helpers ----------------
struct MlpP { const float *b0, *g0, *be0, *b1, *g1, *be1, *b2, *b3; };

__device__ __forceinline__ float lrelu(float x) { return x >= 0.f ? x : 0.1f * x; }

__device__ __forceinline__ v8f wmma_f16(v16h a, v16h b, v8f c) {
    return __builtin_amdgcn_wmma_f32_16x16x32_f16(false, a, false, b, (short)0, c, false, false);
}

union FragCv { uint4 u[2]; v16h h; };

// B fragment (32x16): VGPR v<4 <-> kp2 = kstep*16 + (lane>=16?4:0) + v, VGPR v>=4 <-> kp2 = kstep*16 + 8 + (lane>=16?4:0) + (v-4)
__device__ __forceinline__ v16h load_bfrag(const unsigned* sW, int off, int strideU,
                                           int kstep, int nt, int lane) {
    int n = nt * 16 + (lane & 15);
    const unsigned* p = sW + off + n * strideU + kstep * 16 + ((lane & 16) ? 4 : 0);
    FragCv cv;
    cv.u[0] = *(const uint4*)p;        // VGPR 0..3
    cv.u[1] = *(const uint4*)(p + 8);  // VGPR 4..7
    return cv.h;
}

// A fragment (16x32) from row-major packed-f16 acts: halves k..k+7 and k+16..k+23 contiguous
__device__ __forceinline__ v16h afrag_f16(const unsigned short* acth, int kbase, int lane) {
    int m = lane & 15;
    const unsigned short* p = acth + m * SAH + kbase + ((lane & 16) ? 8 : 0);
    FragCv cv;
    cv.u[0] = *(const uint4*)p;         // K pairs 0..7   (VGPR 0..3)
    cv.u[1] = *(const uint4*)(p + 16);  // K pairs 16..23 (VGPR 4..7)
    return cv.h;
}

// register-resident bias + LayerNorm + LeakyReLU over 8 D-tiles (vector ops on the natural
// accumulator tuples -> v_pk_* without register shuffling), store packed f16 acts
__device__ __forceinline__ void ln_store(v8f* acc, const float* bias, const float* g,
                                         const float* be, unsigned short* acth, int lane) {
    int nlo = lane & 15, rb = (lane & 16) ? 8 : 0;
    v8f ps = {}, pq = {};
#pragma unroll
    for (int nt = 0; nt < 8; ++nt) {
        v8f x = acc[nt] + bias[nt * 16 + nlo];
        acc[nt] = x;
        ps += x;
        pq += x * x;
    }
    v8f mu, rs;
#pragma unroll
    for (int r = 0; r < 8; ++r) {
        float s = ps[r], q = pq[r];
#pragma unroll
        for (int m = 1; m < 16; m <<= 1) {
            s += __shfl_xor(s, m, 32);
            q += __shfl_xor(q, m, 32);
        }
        float mm = s * (1.f / 128.f);
        mu[r] = mm;
        rs[r] = rsqrtf(q * (1.f / 128.f) - mm * mm + 1e-5f);
    }
#pragma unroll
    for (int nt = 0; nt < 8; ++nt) {
        float gg = g[nt * 16 + nlo], bb = be[nt * 16 + nlo];
        v8f x = (acc[nt] - mu) * rs * gg + bb;
#pragma unroll
        for (int r = 0; r < 8; ++r)
            acth[(rb + r) * SAH + nt * 16 + nlo] = f2h_bits(lrelu(x[r]));
    }
}

// ---------------- MLP: 24 ->128(LN,lrelu) ->128(LN,lrelu) ->64(lrelu) ->144 ----------------
__device__ void run_mlp(const unsigned* __restrict__ wgt, MlpP P,
                        const float* __restrict__ hr, const float* __restrict__ hi,
                        int chunkBase, unsigned* sW, unsigned short* sActH,
                        float* sPrm, float* sFout) {
    int tid = threadIdx.x, lane = tid & 31, wave = tid >> 5, nlo = lane & 15;
    int rb = (lane & 16) ? 8 : 0;
    { // weights -> LDS (b128 copies)
        const uint4* src = (const uint4*)wgt;
        uint4* dst = (uint4*)sW;
        for (int i = tid; i < W_TOT / 4; i += THREADS) dst[i] = src[i];
    }
    for (int i = tid; i < 128; i += THREADS) {
        sPrm[i]       = P.b0[i]; sPrm[128 + i] = P.g0[i]; sPrm[256 + i] = P.be0[i];
        sPrm[384 + i] = P.b1[i]; sPrm[512 + i] = P.g1[i]; sPrm[640 + i] = P.be1[i];
    }
    for (int i = tid; i < 64;  i += THREADS) sPrm[768 + i] = P.b2[i];
    for (int i = tid; i < 144; i += THREADS) sPrm[832 + i] = P.b3[i];
    __syncthreads();   // weights/params visible to all waves

    unsigned short* acth = sActH + wave * 16 * SAH;
    const v8f zacc = {};   // inline-0 C operand for WMMA

    // ----- layer 0: X[16x24 (pad32)] @ w0 -----
    {
        int chunk = chunkBase + wave * 16 + nlo;
        int c12 = chunk * 12;
        bool lo = (lane & 16) == 0;
        // lanes<16 need k:{0..7}=re[0..7], {16..23}=im[4..11]
        // lanes>=16 need k:{8..15}=re[8..11],im[0..3], {24..31}=0
        const float4 q1 = *(const float4*)(lo ? hr + c12     : hr + c12 + 8);
        const float4 q2 = *(const float4*)(lo ? hr + c12 + 4 : hi + c12);
        const float4 q3 = *(const float4*)(hi + c12 + 4);
        const float4 q4 = *(const float4*)(hi + c12 + 8);
        v16h a;
        a[0] = (_Float16)q1.x; a[1] = (_Float16)q1.y; a[2] = (_Float16)q1.z; a[3] = (_Float16)q1.w;
        a[4] = (_Float16)q2.x; a[5] = (_Float16)q2.y; a[6] = (_Float16)q2.z; a[7] = (_Float16)q2.w;
        if (lo) {
            a[8]  = (_Float16)q3.x; a[9]  = (_Float16)q3.y; a[10] = (_Float16)q3.z; a[11] = (_Float16)q3.w;
            a[12] = (_Float16)q4.x; a[13] = (_Float16)q4.y; a[14] = (_Float16)q4.z; a[15] = (_Float16)q4.w;
        } else {
#pragma unroll
            for (int e = 8; e < 16; ++e) a[e] = (_Float16)0.f;
        }
        v8f acc[8];
#pragma unroll
        for (int nt = 0; nt < 8; ++nt)
            acc[nt] = wmma_f16(a, load_bfrag(sW, W_L0_OFF, WS_L0, 0, nt, lane), zacc);
        ln_store(acc, sPrm, sPrm + 128, sPrm + 256, acth, lane);
    }
    lds_fence();

    // ----- layer 1: [16x128] @ w1[128x128] -----
    {
        v16h A0 = afrag_f16(acth, 0, lane),  A1 = afrag_f16(acth, 32, lane);
        v16h A2 = afrag_f16(acth, 64, lane), A3 = afrag_f16(acth, 96, lane);
        lds_fence();
        v8f acc[8];
#pragma unroll
        for (int nt = 0; nt < 8; ++nt) {
            acc[nt] = wmma_f16(A0, load_bfrag(sW, W_L1_OFF, WS_L1, 0, nt, lane), zacc);
            acc[nt] = wmma_f16(A1, load_bfrag(sW, W_L1_OFF, WS_L1, 1, nt, lane), acc[nt]);
            acc[nt] = wmma_f16(A2, load_bfrag(sW, W_L1_OFF, WS_L1, 2, nt, lane), acc[nt]);
            acc[nt] = wmma_f16(A3, load_bfrag(sW, W_L1_OFF, WS_L1, 3, nt, lane), acc[nt]);
        }
        ln_store(acc, sPrm + 384, sPrm + 512, sPrm + 640, acth, lane);
    }
    lds_fence();

    // ----- layer 2: [16x128] @ w2[128x64], lrelu -----
    {
        v16h A0 = afrag_f16(acth, 0, lane),  A1 = afrag_f16(acth, 32, lane);
        v16h A2 = afrag_f16(acth, 64, lane), A3 = afrag_f16(acth, 96, lane);
        lds_fence();
#pragma unroll
        for (int nt = 0; nt < 4; ++nt) {
            v8f acc = wmma_f16(A0, load_bfrag(sW, W_L2_OFF, WS_L2, 0, nt, lane), zacc);
            acc = wmma_f16(A1, load_bfrag(sW, W_L2_OFF, WS_L2, 1, nt, lane), acc);
            acc = wmma_f16(A2, load_bfrag(sW, W_L2_OFF, WS_L2, 2, nt, lane), acc);
            acc = wmma_f16(A3, load_bfrag(sW, W_L2_OFF, WS_L2, 3, nt, lane), acc);
            v8f y = acc + sPrm[768 + nt * 16 + nlo];
#pragma unroll
            for (int r = 0; r < 8; ++r)
                acth[(rb + r) * SAH + nt * 16 + nlo] = f2h_bits(lrelu(y[r]));
        }
    }
    lds_fence();

    // ----- layer 3: [16x64] @ w3[64x144] -> f32 factors in LDS -----
    {
        v16h A0 = afrag_f16(acth, 0, lane), A1 = afrag_f16(acth, 32, lane);
        float* fo = sFout + wave * 16 * FS;
#pragma unroll
        for (int nt = 0; nt < 9; ++nt) {
            v8f acc = wmma_f16(A0, load_bfrag(sW, W_L3_OFF, WS_L3, 0, nt, lane), zacc);
            acc = wmma_f16(A1, load_bfrag(sW, W_L3_OFF, WS_L3, 1, nt, lane), acc);
            v8f y = acc + sPrm[832 + nt * 16 + nlo];
            int n = nt * 16 + nlo;
#pragma unroll
            for (int r = 0; r < 8; ++r) fo[(rb + r) * FS + n] = y[r];
        }
    }
    __syncthreads();   // factors visible to all waves; sW free for reuse
}

// ---------------- complex helpers ----------------
struct cplx { float re, im; };
__device__ __forceinline__ cplx cmul(cplx a, cplx b) { return { a.re*b.re - a.im*b.im, a.re*b.im + a.im*b.re }; }
__device__ __forceinline__ cplx csub(cplx a, cplx b) { return { a.re - b.re, a.im - b.im }; }
__device__ __forceinline__ cplx cdiv(cplx a, cplx b) {
    float d = 1.f / (b.re*b.re + b.im*b.im);
    return { (a.re*b.re + a.im*b.im) * d, (a.im*b.re - a.re*b.im) * d };
}
__device__ __forceinline__ float softplus(float x) {
    return (x > 0.f) ? x + log1pf(__expf(-x)) : log1pf(__expf(x));
}

// ---------------- fused mega-kernel ----------------
__global__ __launch_bounds__(THREADS)
void mmse_mega(const float* __restrict__ hr, const float* __restrict__ hi,
               const unsigned* __restrict__ wgtC, const unsigned* __restrict__ wgtR,
               MlpP pC, MlpP pR, float* __restrict__ out) {
    extern __shared__ char smem[];
    unsigned* sW        = (unsigned*)(smem + LDS_W_B);
    unsigned short* sAH = (unsigned short*)(smem + LDS_ACT_B);
    float* sFC          = (float*)(smem + LDS_FC_B);
    float* sFR          = (float*)(smem + LDS_FR_B);
    float* sPrm         = (float*)(smem + LDS_PRM_B);

    // prefetch R-MLP weights into cache while C-MLP runs
    for (int i = threadIdx.x * 128; i < W_TOT * 4; i += THREADS * 128)
        __builtin_prefetch((const char*)wgtR + i, 0, 3);

    int chunkBase = blockIdx.x * ROWS_WG;
    run_mlp(wgtC, pC, hr, hi, chunkBase, sW, sAH, sPrm, sFC);
    run_mlp(wgtR, pR, hr, hi, chunkBase, sW, sAH, sPrm, sFR);

    // ---- post: build C,R (L@L^H + eps I), solve (C+R)x = chunk, out = C x ----
    // All solver communication is intra-wave (16-lane groups = half-waves); same-wave LDS
    // ops are in-order on CDNA5, so compiler fences replace block barriers.
    // Branchless: lanes 12-15 duplicate row 11's work (ii clamp) into private scratch rows,
    // avoiding exec-mask save/restore; only shared writes stay guarded.
    float* sL   = (float*)smem;            // 16 groups * 16 rows * 24 floats (reuses sW)
    float* sPiv = sL + 16 * 16 * 24;       // 16*12*26 floats
    float* sX   = sPiv + 16 * 12 * 26;     // 16*12*2 floats
    int tid = threadIdx.x;
    int gg = tid >> 4;                     // chunk group (0..15); groups 2g,2g+1 in wave g
    int i  = tid & 15;                     // lane row; rows 0..11 are real
    int ii = i < 12 ? i : 11;              // clamped data index (dup of row 11 for 12..15)
    float* myL = sL + (gg * 16 + i) * 24;  // private scratch row per lane

    for (int batch = 0; batch < 8; ++batch) {
        int cl = batch * 16 + gg;
        int n  = chunkBase + cl;
        const float* fC = sFC + cl * FS;
        const float* fR = sFR + cl * FS;
        cplx Crow[12], Arow[12];
        int tb = ii * (ii - 1) / 2;
        const float* Lg = sL + gg * 16 * 24;

        // Lc row
        {
#pragma unroll
            for (int j = 0; j < 12; ++j) { myL[2*j] = 0.f; myL[2*j+1] = 0.f; }
            for (int j = 0; j < ii; ++j) { myL[2*j] = fC[12 + tb + j]; myL[2*j+1] = fC[78 + tb + j]; }
            myL[2*ii] = softplus(fC[ii]);
        }
        lds_fence();
        {                                   // C[i][j] = sum_k Lc[i,k] conj(Lc[j,k])
#pragma unroll
            for (int j = 0; j < 12; ++j) {
                cplx s = { 0.f, 0.f };
                int mn = ii < j ? ii : j;
                for (int k = 0; k <= mn; ++k) {
                    float ar = myL[2*k], ai = myL[2*k+1];
                    float br = Lg[j*24 + 2*k], bi = Lg[j*24 + 2*k + 1];
                    s.re += ar * br + ai * bi;
                    s.im += ai * br - ar * bi;
                }
                Crow[j] = s;
            }
            Crow[ii].re += 1e-6f;
        }
        lds_fence();
        // Lr row
        {
#pragma unroll
            for (int j = 0; j < 12; ++j) { myL[2*j] = 0.f; myL[2*j+1] = 0.f; }
            for (int j = 0; j < ii; ++j) { myL[2*j] = fR[12 + tb + j]; myL[2*j+1] = fR[78 + tb + j]; }
            myL[2*ii] = softplus(fR[ii]);
        }
        lds_fence();
        {                                   // A = C + R
#pragma unroll
            for (int j = 0; j < 12; ++j) {
                cplx s = { 0.f, 0.f };
                int mn = ii < j ? ii : j;
                for (int k = 0; k <= mn; ++k) {
                    float ar = myL[2*k], ai = myL[2*k+1];
                    float br = Lg[j*24 + 2*k], bi = Lg[j*24 + 2*k + 1];
                    s.re += ar * br + ai * bi;
                    s.im += ai * br - ar * bi;
                }
                Arow[j].re = Crow[j].re + s.re;
                Arow[j].im = Crow[j].im + s.im;
            }
            Arow[ii].re += 1e-6f;           // R's eps
        }
        lds_fence();

        cplx b = { hr[n * 12 + ii], hi[n * 12 + ii] };

        // forward elimination (A is HPD -> no pivoting)
        float* gP = sPiv + gg * 12 * 26;
        for (int k = 0; k < 12; ++k) {
            if (i == k) {
#pragma unroll
                for (int j = 0; j < 12; ++j) { gP[k*26 + 2*j] = Arow[j].re; gP[k*26 + 2*j + 1] = Arow[j].im; }
                gP[k*26 + 24] = b.re; gP[k*26 + 25] = b.im;
            }
            lds_fence();
            if (i > k) {
                cplx piv = { gP[k*26 + 2*k], gP[k*26 + 2*k + 1] };
                cplx f = cdiv(Arow[k], piv);
                for (int j = k; j < 12; ++j) {
                    cplx p = { gP[k*26 + 2*j], gP[k*26 + 2*j + 1] };
                    Arow[j] = csub(Arow[j], cmul(f, p));
                }
                cplx pb = { gP[k*26 + 24], gP[k*26 + 25] };
                b = csub(b, cmul(f, pb));
            }
            lds_fence();
        }
        // back substitution
        for (int k = 11; k >= 0; --k) {
            if (i == k) {
                cplx x = cdiv(b, Arow[k]);
                sX[(gg*12 + k)*2] = x.re; sX[(gg*12 + k)*2 + 1] = x.im;
            }
            lds_fence();
            if (i < k) {
                cplx xk = { sX[(gg*12 + k)*2], sX[(gg*12 + k)*2 + 1] };
                b = csub(b, cmul(Arow[k], xk));
            }
        }
        lds_fence();
        // out_i = sum_j C[i][j] * x[j]
        if (i < 12) {
            cplx o = { 0.f, 0.f };
#pragma unroll
            for (int j = 0; j < 12; ++j) {
                cplx xj = { sX[(gg*12 + j)*2], sX[(gg*12 + j)*2 + 1] };
                cplx t = cmul(Crow[j], xj);
                o.re += t.re; o.im += t.im;
            }
            int p = n * 12 + i;
            out[2*p] = o.re; out[2*p + 1] = o.im;
        }
        lds_fence();
    }
}

// ---------------- host launcher ----------------
extern "C" void kernel_launch(void* const* d_in, const int* in_sizes, int n_in,
                              void* d_out, int out_size, void* d_ws, size_t ws_size,
                              hipStream_t stream) {
    // Detect input flattening: top-level insertion vs pytree-sorted; nested dict order.
    const float *hr, *hi;
    int cbase, rbase;
    bool topFirst = (n_in >= 2 && in_sizes[0] > 24 * 128 * 2); // h_real has 1.57M elems
    if (topFirst) { hr = (const float*)d_in[0]; hi = (const float*)d_in[1]; cbase = 2; rbase = 14; }
    else          { cbase = 0; rbase = 12; hi = (const float*)d_in[24]; hr = (const float*)d_in[25]; }
    // canonical order: w0,b0,w1,b1,w2,b2,w3,b3,g0,be0,g1,be1
    // sorted leaf order: b0,b1,b2,b3,be0,be1,g0,g1,w0,w1,w2,w3
    const int ms[12] = { 8, 0, 9, 1, 10, 2, 11, 3, 6, 4, 7, 5 };
    bool sortedLeaves = (in_sizes[cbase] != 24 * 128);  // insertion order starts with w0 (3072)
    const float *C[12], *R[12];
    for (int k = 0; k < 12; ++k) {
        int off = sortedLeaves ? ms[k] : k;
        C[k] = (const float*)d_in[cbase + off];
        R[k] = (const float*)d_in[rbase + off];
    }

    unsigned* wgtC = (unsigned*)d_ws;
    unsigned* wgtR = wgtC + W_TOT;

    // pack weights (fp32 -> K-major padded pair-packed f16)
    struct { const float* w; int Kr, strideU, N; unsigned* o; } jobs[8] = {
        { C[0], 24,  WS_L0, 128, wgtC + W_L0_OFF }, { C[2], 128, WS_L1, 128, wgtC + W_L1_OFF },
        { C[4], 128, WS_L2, 64,  wgtC + W_L2_OFF }, { C[6], 64,  WS_L3, 144, wgtC + W_L3_OFF },
        { R[0], 24,  WS_L0, 128, wgtR + W_L0_OFF }, { R[2], 128, WS_L1, 128, wgtR + W_L1_OFF },
        { R[4], 128, WS_L2, 64,  wgtR + W_L2_OFF }, { R[6], 64,  WS_L3, 144, wgtR + W_L3_OFF },
    };
    for (int j = 0; j < 8; ++j) {
        int tot = jobs[j].N * jobs[j].strideU;
        pack_weights<<<(tot + 255) / 256, 256, 0, stream>>>(jobs[j].w, jobs[j].Kr,
                                                            jobs[j].strideU, jobs[j].N, jobs[j].o);
    }

    MlpP pC = { C[1], C[8], C[9], C[3], C[10], C[11], C[5], C[7] };
    MlpP pR = { R[1], R[8], R[9], R[3], R[10], R[11], R[5], R[7] };

    int nchunks = (topFirst ? in_sizes[0] : in_sizes[25]) / 12;   // 131072
    int grid = nchunks / ROWS_WG;                                  // 1024

    (void)hipFuncSetAttribute((const void*)mmse_mega,
                              hipFuncAttributeMaxDynamicSharedMemorySize, LDS_TOT_B);
    mmse_mega<<<grid, THREADS, LDS_TOT_B, stream>>>(hr, hi, wgtC, wgtR, pC, pR, (float*)d_out);
    (void)out_size; (void)ws_size;
}
